// VI_TDCN_20014547599684
// MI455X (gfx1250) — compile-verified, matrix operand order checked by reading, not run on hardware
//
#include <hip/hip_runtime.h>
#include <hip/hip_bf16.h>

// ---------------------------------------------------------------------------
// VI-TDCN forward for gfx1250 (MI455X).
// 3x3 convs = 9 shifted 1x1 GEMMs on V_WMMA_F32_16X16X32_F16:
//   each K-chunk = 32 input channels at one tap; LDS tile stored channel-major
//   so B fragments are two ds_load_b128 with immediate tap offsets.
// Weights are device-prepacked into the exact WMMA A-fragment layout.
// ---------------------------------------------------------------------------

typedef __attribute__((ext_vector_type(16))) _Float16 v16h;
typedef __attribute__((ext_vector_type(8)))  _Float16 v8h;
typedef __attribute__((ext_vector_type(8)))  float    v8f;

#define H2 128
#define W2 128
#define HW2 (H2 * W2)
#define CSTR 40                 // LDS per-pixel channel-run stride in halves (80 B)

__device__ __forceinline__ float lrelu_f(float v) { return v >= 0.f ? v : 0.2f * v; }

// ---------------------------------------------------------------------------
// Weight pre-pack: f32 [Cout, Cin, 3, 3] -> f16 WMMA A-fragments.
// K-chunk order: (channel-group g, tap). Within a chunk, 16-bit A layout:
//   lane%16 = M, element e -> k = (e&7) + ((e>>3)<<4) + ((lane>>4)<<3)
// Packed index: (((mtile*NKC + chunk)*32 + lane)*16 + e).
// ---------------------------------------------------------------------------
__global__ void pack_w(const float* __restrict__ w, _Float16* __restrict__ ap,
                       int Cout, int Cin, int mtiles) {
    int nkc = (Cin / 32) * 9;
    long total = (long)mtiles * nkc * 512;
    long idx = (long)blockIdx.x * 256 + threadIdx.x;
    if (idx >= total) return;
    int e    = idx & 15;
    int lane = (idx >> 4) & 31;
    long rest = idx >> 9;                 // mt*nkc + chunk
    int kc = (int)(rest % nkc);
    int mt = (int)(rest / nkc);
    int g   = kc / 9;
    int tap = kc % 9;
    int m = mt * 16 + (lane & 15);
    int c = g * 32 + (e & 7) + ((e >> 3) << 4) + ((lane >> 4) << 3);
    float v = 0.f;
    if (m < Cout) v = w[((long)m * Cin + c) * 9 + tap];
    ap[idx] = (_Float16)v;
}

// ---------------------------------------------------------------------------
// Implicit-GEMM 3x3 conv (pad 1, stride 1), WMMA f16 -> f32.
// Block: 256 thr = 8 waves, 4(M) x 2(pixel rows). Output tile: 64 Cout x
// (16 wide x 2 tall). Channels staged 32 at a time into LDS channel-major
// with halo; 9 taps -> 9 WMMAs per channel group.
// Two input pointers + csplit implement channel concatenation for free.
// ---------------------------------------------------------------------------
__global__ __launch_bounds__(256) void conv3x3_wmma(
    const _Float16* __restrict__ in0, const _Float16* __restrict__ in1,
    int csplit, int ngroups, long bstride0, long bstride1,
    const _Float16* __restrict__ apack, const float* __restrict__ bias,
    int Cout, int mblocks,
    float* __restrict__ outf, _Float16* __restrict__ outh,
    long obstride, int relu)
{
    // [4 halo rows][18 halo cols][40 (32ch + pad)] halves, 16B-aligned runs
    __shared__ __align__(16) _Float16 smem[4 * 18 * CSTR];

    const int tid  = threadIdx.x;
    const int lane = tid & 31;
    const int wave = tid >> 5;
    const int wm   = wave & 3;     // which 16-row Cout tile
    const int wn   = wave >> 2;    // which pixel row of the 2-row tile
    const int x0   = blockIdx.x * 16;
    const int y0   = blockIdx.y * 2;
    const int b    = blockIdx.z / mblocks;
    const int mb   = blockIdx.z % mblocks;
    const int nkc  = ngroups * 9;
    const int n    = lane & 15;                // N (pixel col) within wave tile
    const int khalf = (lane >> 4) << 3;        // K-half select (in halves)

    const _Float16* i0 = in0 + (long)b * bstride0;
    const _Float16* i1 = in1 + (long)b * bstride1;

    // per-lane B base: row (wn), col (n), channel offset khalf
    const _Float16* bbase = smem + ((wn * 18 + n) * CSTR + khalf);
    const _Float16* abase = apack + ((long)(mb * 4 + wm) * nkc << 5) * 16 + lane * 16;

    v8f acc = {};

    for (int g = 0; g < ngroups; ++g) {
        // ---- stage 32 channels with 3x3 halo, channel-major in LDS ----
        // i: col fastest (coalesced global reads), store strided into LDS
        for (int i = tid; i < 32 * 4 * 18; i += 256) {
            int col = i % 18;
            int r   = (i / 18) & 3;
            int cl  = i / 72;
            int y = y0 - 1 + r;
            int x = x0 - 1 + col;
            int cg = (g << 5) + cl;
            _Float16 v = (_Float16)0.f;
            if ((unsigned)y < (unsigned)H2 && (unsigned)x < (unsigned)W2) {
                const _Float16* src = (cg < csplit)
                    ? (i0 + (long)cg * HW2)
                    : (i1 + (long)(cg - csplit) * HW2);
                v = src[y * W2 + x];
            }
            smem[(r * 18 + col) * CSTR + cl] = v;
        }
        __syncthreads();

        // ---- 9 taps: each is one 16x16x32 WMMA over 32 channels ----
        const _Float16* ag = abase + ((long)g * 9 << 5) * 16;
#pragma unroll
        for (int tap = 0; tap < 9; ++tap) {
            const int dy = tap / 3, dx = tap % 3;          // compile-time
            const v16h a = *(const v16h*)(ag + ((long)tap << 5) * 16);
            const _Float16* p = bbase + (dy * 18 + dx) * CSTR;
            v8h lo = *(const v8h*)(p);        // channels khalf..khalf+7
            v8h hi = *(const v8h*)(p + 16);   // channels khalf+16..khalf+23
            v16h bfr = __builtin_shufflevector(lo, hi,
                0, 1, 2, 3, 4, 5, 6, 7, 8, 9, 10, 11, 12, 13, 14, 15);
            acc = __builtin_amdgcn_wmma_f32_16x16x32_f16(
                false, a, false, bfr, (short)0, acc, false, false);
        }
        __syncthreads();
    }

    // ---- epilogue: bias + optional LeakyReLU, dual f32/f16 stores ----
    const int mlo = mb * 64 + wm * 16 + ((lane >> 4) << 3);
    const int ny = y0 + wn;
    const int nx = x0 + n;
    const long pix = (long)ny * W2 + nx;
#pragma unroll
    for (int r = 0; r < 8; ++r) {
        int m = mlo + r;
        if (m < Cout) {
            float v = acc[r] + bias[m];
            if (relu) v = lrelu_f(v);
            long o = (long)b * obstride + (long)m * HW2 + pix;
            if (outf) outf[o] = v;
            if (outh) outh[o] = (_Float16)v;
        }
    }
}

// ---------------------------------------------------------------------------
// 1x1x1 channel GEMM (128->64) + nearest 2x upsample.
// Writes xt f32 [B,T,64,128,128] and refcat f16 channels [0,64).
// ---------------------------------------------------------------------------
__global__ void conv_trans_up(const float* __restrict__ x, const float* __restrict__ w,
                              const float* __restrict__ bs, float* __restrict__ xtf,
                              _Float16* __restrict__ refcat)
{
    int p  = blockIdx.x * 256 + threadIdx.x;   // 0..4095 over 64x64
    int o  = blockIdx.y;
    int bt = blockIdx.z;
    int b = bt / 3, t = bt % 3;
    float acc = bs[o];
    const float* xp = x + (((long)b * 128) * 3 + t) * 4096 + p;
    for (int c = 0; c < 128; ++c)
        acc += w[o * 128 + c] * xp[(long)c * 3 * 4096];
    int h = p >> 6, ww = p & 63;
    long base = (((long)bt * 64 + o) * HW2) + (long)(h * 2) * W2 + ww * 2;
    xtf[base] = acc; xtf[base + 1] = acc; xtf[base + W2] = acc; xtf[base + W2 + 1] = acc;
    long rb = (((long)bt * 192 + o) * HW2) + (long)(h * 2) * W2 + ww * 2;
    _Float16 hv = (_Float16)acc;
    refcat[rb] = hv; refcat[rb + 1] = hv; refcat[rb + W2] = hv; refcat[rb + W2 + 1] = hv;
}

// refcat channels [64,192): f32 vfeat/ifeat -> f16
__global__ void fill_refcat(const float* __restrict__ vf, const float* __restrict__ iff,
                            _Float16* __restrict__ refcat)
{
    long idx = (long)blockIdx.x * 256 + threadIdx.x;
    if (idx >= 6L * 128 * HW2) return;
    int p   = (int)(idx % HW2);
    long r  = idx / HW2;
    int cc  = (int)(r % 128);
    int bt  = (int)(r / 128);
    const float* src = (cc < 64) ? (vf + ((long)bt * 64 + cc) * HW2)
                                 : (iff + ((long)bt * 64 + cc - 64) * HW2);
    refcat[((long)bt * 192 + 64 + cc) * HW2 + p] = (_Float16)src[p];
}

// ---------------------------------------------------------------------------
// Deformable 3x3 conv, 8 deform groups == 8 weight groups, cpg = 8.
// One thread: one (b, g, pixel), produces 8 output channels.
// ---------------------------------------------------------------------------
__global__ __launch_bounds__(256) void deform_conv_k(
    const _Float16* __restrict__ nfea, const float* __restrict__ off,
    const float* __restrict__ dw, const float* __restrict__ db,
    _Float16* __restrict__ out)
{
    __shared__ float wl[576];   // [oo][c][tap] for this group
    __shared__ float bl[8];
    int tid = threadIdx.x;
    int g = blockIdx.y, b = blockIdx.z;
    for (int i = tid; i < 576; i += 256) wl[i] = dw[(long)g * 576 + i];
    if (tid < 8) bl[tid] = db[g * 8 + tid];
    __syncthreads();

    int p = blockIdx.x * 256 + tid;
    int y = p >> 7, x = p & 127;
    float acc[8];
#pragma unroll
    for (int oo = 0; oo < 8; ++oo) acc[oo] = bl[oo];

    const float*    offb = off  + ((long)b * 144) * HW2;
    const _Float16* nb   = nfea + ((long)b * 64 + g * 8) * HW2;

    for (int tap = 0; tap < 9; ++tap) {
        int ky = tap / 3, kx = tap % 3;
        float oy = offb[((long)((g * 9 + tap) * 2 + 0)) * HW2 + p];
        float ox = offb[((long)((g * 9 + tap) * 2 + 1)) * HW2 + p];
        float py = (float)(y + ky - 1) + oy;
        float px = (float)(x + kx - 1) + ox;
        float fy = floorf(py), fx = floorf(px);
        float wy = py - fy, wx = px - fx;
        int yi = (int)fy, xi = (int)fx;
        float cw[4] = { (1.f - wy) * (1.f - wx), (1.f - wy) * wx,
                        wy * (1.f - wx),         wy * wx };
        int idxs[4];
#pragma unroll
        for (int cor = 0; cor < 4; ++cor) {
            int yy = yi + (cor >> 1), xx = xi + (cor & 1);
            bool ok = (yy >= 0) && (yy < H2) && (xx >= 0) && (xx < W2);
            int yc = min(max(yy, 0), H2 - 1);
            int xc = min(max(xx, 0), W2 - 1);
            idxs[cor] = yc * W2 + xc;
            if (!ok) cw[cor] = 0.f;
        }
#pragma unroll
        for (int c = 0; c < 8; ++c) {
            const _Float16* ch = nb + (long)c * HW2;
            float v = cw[0] * (float)ch[idxs[0]] + cw[1] * (float)ch[idxs[1]]
                    + cw[2] * (float)ch[idxs[2]] + cw[3] * (float)ch[idxs[3]];
#pragma unroll
            for (int oo = 0; oo < 8; ++oo)
                acc[oo] += v * wl[(oo * 8 + c) * 9 + tap];
        }
    }
    _Float16* ob = out + ((long)b * 64 + g * 8) * HW2 + p;
#pragma unroll
    for (int oo = 0; oo < 8; ++oo) ob[(long)oo * HW2] = (_Float16)acc[oo];
}

// ---------------------------------------------------------------------------
// Temporal (3,1,1) conv + LeakyReLU + residual add. Writes enhanced
// [B,64,T,H2,W2] f32 into d_out. Weight index uniform per block -> s_loads.
// ---------------------------------------------------------------------------
__global__ void temporal_out(const _Float16* __restrict__ h1, const float* __restrict__ tw,
                             const float* __restrict__ tb, const float* __restrict__ xtf,
                             float* __restrict__ enh)
{
    int p  = blockIdx.x * 256 + threadIdx.x;
    int o  = blockIdx.y;
    int bt = blockIdx.z;
    int b = bt / 3, t = bt % 3;
    float acc = tb[o];
    for (int dt = 0; dt < 3; ++dt) {
        int tt = t + dt - 1;
        if (tt < 0 || tt >= 3) continue;
        const _Float16* hp = h1 + ((long)(b * 3 + tt) * 64) * HW2 + p;
        const float* wp = tw + (long)o * 64 * 3 + dt;
        for (int c = 0; c < 64; ++c)
            acc += wp[c * 3] * (float)hp[(long)c * HW2];
    }
    float v = lrelu_f(acc) + xtf[((long)(b * 3 + t) * 64 + o) * HW2 + p];
    enh[(((long)b * 64 + o) * 3 + t) * HW2 + p] = v;
}

// ---------------------------------------------------------------------------
extern "C" void kernel_launch(void* const* d_in, const int* in_sizes, int n_in,
                              void* d_out, int out_size, void* d_ws, size_t ws_size,
                              hipStream_t stream)
{
    (void)in_sizes; (void)n_in; (void)out_size; (void)ws_size;
    const float* x     = (const float*)d_in[0];
    const float* vfeat = (const float*)d_in[1];
    const float* ifeat = (const float*)d_in[2];
    int pi = 3;
    const float* ct_w   = (const float*)d_in[pi++]; const float* ct_b   = (const float*)d_in[pi++];
    const float* cm_w   = (const float*)d_in[pi++]; const float* cm_b   = (const float*)d_in[pi++];
    const float* cf_w   = (const float*)d_in[pi++]; const float* cf_b   = (const float*)d_in[pi++];
    const float* cf2_w  = (const float*)d_in[pi++]; const float* cf2_b  = (const float*)d_in[pi++];
    const float* tc_w   = (const float*)d_in[pi++]; const float* tc_b   = (const float*)d_in[pi++];
    const float* tc2_w  = (const float*)d_in[pi++]; const float* tc2_b  = (const float*)d_in[pi++];
    const float* bo_w   = (const float*)d_in[pi++]; const float* bo_b   = (const float*)d_in[pi++];
    const float* bo2_w  = (const float*)d_in[pi++]; const float* bo2_b  = (const float*)d_in[pi++];
    const float* off1_w = (const float*)d_in[pi++]; const float* off1_b = (const float*)d_in[pi++];
    const float* off2_w = (const float*)d_in[pi++]; const float* off2_b = (const float*)d_in[pi++];
    const float* d1_w   = (const float*)d_in[pi++]; const float* d1_b   = (const float*)d_in[pi++];
    const float* d2_w   = (const float*)d_in[pi++]; const float* d2_b   = (const float*)d_in[pi++];
    const float* fn_w   = (const float*)d_in[pi++]; const float* fn_b   = (const float*)d_in[pi++];
    const float* sp_w   = (const float*)d_in[pi++]; const float* sp_b   = (const float*)d_in[pi++];
    const float* tm_w   = (const float*)d_in[pi++]; const float* tm_b   = (const float*)d_in[pi++];

    float* enh  = (float*)d_out;                          // [B,64,T,H2,W2]
    float* offs = enh + (size_t)2 * 64 * 3 * HW2;         // [T,2,B,144,H2,W2]

    // ---- workspace arena ----
    size_t woff = 0;
    auto alloc = [&](size_t bytes) -> void* {
        void* p = (char*)d_ws + woff;
        woff += (bytes + 255) & ~(size_t)255;
        return p;
    };
    float*    xtf    = (float*)   alloc(6ull * 64  * HW2 * 4);   // [B,T,64,HW2]
    _Float16* refcat = (_Float16*)alloc(6ull * 192 * HW2 * 2);   // [B,T,192,HW2]
    _Float16* fea    = (_Float16*)alloc(2ull * 64 * HW2 * 2);
    _Float16* nfea1  = (_Float16*)alloc(2ull * 64 * HW2 * 2);
    _Float16* nfea2  = (_Float16*)alloc(2ull * 64 * HW2 * 2);
    _Float16* ft1    = (_Float16*)alloc(2ull * 64 * HW2 * 2);
    _Float16* cfea1  = (_Float16*)alloc(2ull * 64 * HW2 * 2);
    _Float16* fea1   = (_Float16*)alloc(2ull * 64 * HW2 * 2);
    _Float16* ft2    = (_Float16*)alloc(2ull * 64 * HW2 * 2);
    _Float16* cfea2  = (_Float16*)alloc(2ull * 64 * HW2 * 2);
    _Float16* fea2   = (_Float16*)alloc(2ull * 64 * HW2 * 2);
    _Float16* alignd = (_Float16*)alloc(6ull * 64 * HW2 * 2);    // [B,T,64,HW2]
    _Float16* h1buf  = (_Float16*)alloc(6ull * 64 * HW2 * 2);    // [B,T,64,HW2]

    auto packsz = [](int Cout, int Cin) -> size_t {
        int mt = ((Cout + 63) / 64) * 4;
        return (size_t)mt * ((Cin / 32) * 9) * 512;
    };
    _Float16* ap_cm  = (_Float16*)alloc(packsz(64, 192) * 2);
    _Float16* ap_cf  = (_Float16*)alloc(packsz(64, 192) * 2);
    _Float16* ap_cf2 = (_Float16*)alloc(packsz(64, 192) * 2);
    _Float16* ap_tc  = (_Float16*)alloc(packsz(64, 128) * 2);
    _Float16* ap_tc2 = (_Float16*)alloc(packsz(64, 128) * 2);
    _Float16* ap_bo  = (_Float16*)alloc(packsz(64, 64)  * 2);
    _Float16* ap_bo2 = (_Float16*)alloc(packsz(64, 64)  * 2);
    _Float16* ap_o1  = (_Float16*)alloc(packsz(144, 64) * 2);
    _Float16* ap_o2  = (_Float16*)alloc(packsz(144, 64) * 2);
    _Float16* ap_fn  = (_Float16*)alloc(packsz(64, 128) * 2);
    _Float16* ap_sp  = (_Float16*)alloc(packsz(64, 128) * 2);

    auto pack = [&](const float* w, _Float16* ap, int Cout, int Cin) {
        int mt = ((Cout + 63) / 64) * 4;
        long total = (long)mt * ((Cin / 32) * 9) * 512;
        pack_w<<<dim3((unsigned)((total + 255) / 256)), 256, 0, stream>>>(w, ap, Cout, Cin, mt);
    };
    pack(cm_w,  ap_cm,  64, 192);
    pack(cf_w,  ap_cf,  64, 192);
    pack(cf2_w, ap_cf2, 64, 192);
    pack(tc_w,  ap_tc,  64, 128);
    pack(tc2_w, ap_tc2, 64, 128);
    pack(bo_w,  ap_bo,  64, 64);
    pack(bo2_w, ap_bo2, 64, 64);
    pack(off1_w, ap_o1, 144, 64);
    pack(off2_w, ap_o2, 144, 64);
    pack(fn_w,  ap_fn,  64, 128);
    pack(sp_w,  ap_sp,  64, 128);

    auto conv = [&](const _Float16* in0, const _Float16* in1, int csplit, int Cin,
                    long bs0, long bs1, const _Float16* ap, const float* bias, int Cout,
                    float* outf, _Float16* outh, long obs, int relu) {
        int mblocks = (Cout + 63) / 64;
        dim3 grid(W2 / 16, H2 / 2, 2 * mblocks);
        conv3x3_wmma<<<grid, 256, 0, stream>>>(in0, in1, csplit, Cin / 32, bs0, bs1,
                                               ap, bias, Cout, mblocks, outf, outh, obs, relu);
    };

    // xt (1x1 GEMM) + upsample, then remaining refcat channels
    conv_trans_up<<<dim3(16, 64, 6), 256, 0, stream>>>(x, ct_w, ct_b, xtf, refcat);
    fill_refcat<<<dim3((unsigned)((6L * 128 * HW2 + 255) / 256)), 256, 0, stream>>>(vfeat, ifeat, refcat);

    const long RCB  = 3L * 192 * HW2;   // refcat batch stride
    const long BS64 = 64L * HW2;        // [B,64,HW2] batch stride
    for (int t = 0; t < 3; ++t) {
        const _Float16* rc_t = refcat + (long)t * 192 * HW2;
        int n1 = (t < 2) ? t : t - 1;
        int n2 = (t < 2) ? t : t - 2;
        const _Float16* rc_n1 = refcat + (long)n1 * 192 * HW2;
        const _Float16* rc_n2 = refcat + (long)n2 * 192 * HW2;

        conv(rc_t, rc_t, 192, 192, RCB, RCB, ap_cm, cm_b, 64, nullptr, fea, BS64, 1);

        // path 1
        conv(rc_n1, rc_n1, 192, 192, RCB, RCB, ap_cf, cf_b, 64, nullptr, nfea1, BS64, 1);
        conv(fea, nfea1, 64, 128, BS64, BS64, ap_tc, tc_b, 64, nullptr, ft1, BS64, 1);
        conv(ft1, ft1, 64, 64, BS64, BS64, ap_bo, bo_b, 64, nullptr, cfea1, BS64, 0);
        float* off1p = offs + ((long)t * 2 + 0) * 2 * 144 * HW2;
        conv(cfea1, cfea1, 64, 64, BS64, BS64, ap_o1, off1_b, 144, off1p, nullptr, 144L * HW2, 0);
        deform_conv_k<<<dim3(HW2 / 256, 8, 2), 256, 0, stream>>>(nfea1, off1p, d1_w, d1_b, fea1);

        // path 2
        conv(rc_n2, rc_n2, 192, 192, RCB, RCB, ap_cf2, cf2_b, 64, nullptr, nfea2, BS64, 1);
        conv(fea, nfea2, 64, 128, BS64, BS64, ap_tc2, tc2_b, 64, nullptr, ft2, BS64, 1);
        conv(ft2, ft2, 64, 64, BS64, BS64, ap_bo2, bo2_b, 64, nullptr, cfea2, BS64, 0);
        float* off2p = offs + ((long)t * 2 + 1) * 2 * 144 * HW2;
        conv(cfea2, cfea2, 64, 64, BS64, BS64, ap_o2, off2_b, 144, off2p, nullptr, 144L * HW2, 0);
        deform_conv_k<<<dim3(HW2 / 256, 8, 2), 256, 0, stream>>>(nfea2, off2p, d2_w, d2_b, fea2);

        // fusion conv -> aligned[t]
        conv(fea1, fea2, 64, 128, BS64, BS64, ap_fn, fn_b, 64,
             nullptr, alignd + (long)t * 64 * HW2, 3L * 64 * HW2, 1);
    }

    // Separable 3D: spatial (1,3,3) over cat(xt, aligned) per t
    for (int t = 0; t < 3; ++t) {
        conv(refcat + (long)t * 192 * HW2, alignd + (long)t * 64 * HW2, 64, 128,
             RCB, 3L * 64 * HW2, ap_sp, sp_b, 64,
             nullptr, h1buf + (long)t * 64 * HW2, 3L * 64 * HW2, 1);
    }

    // temporal (3,1,1) + lrelu + residual
    temporal_out<<<dim3(HW2 / 256, 64, 6), 256, 0, stream>>>(h1buf, tm_w, tm_b, xtf, enh);
}
